// DEC_63041529971265
// MI455X (gfx1250) — matrix operand order repaired and checked by reading, not compiled, not run on hardware
//
#include <hip/hip_runtime.h>
#include <cstdint>
#include <cstddef>

typedef __attribute__((ext_vector_type(16))) _Float16 v16h;
typedef __attribute__((ext_vector_type(8)))  _Float16 v8h;
typedef __attribute__((ext_vector_type(8)))  float    v8f;

#define DEV __device__ __forceinline__

namespace {
constexpr int B_   = 256;
constexpr int T_   = 64;
constexpr int H_   = 256;
constexpr int DLY  = 8;
constexpr int ROWS = 16;              // batch rows per block (= WMMA M)
constexpr int NBLK = B_ / ROWS;       // 16 row-groups per stream
constexpr int G3   = 3 * H_;          // 768 gate outputs
constexpr int NW   = 16;              // waves per block
constexpr int THREADS = NW * 32;      // 512
constexpr int H2   = 2 * H_;          // 512

// ---- dynamic LDS layout (bytes) ----
constexpr int OFF_HA   = 0;                               // f32 [2][ROWS][H]  post-attn hidden
constexpr int OFF_ES   = OFF_HA   + 2*ROWS*H_*4;          // f32 [2][ROWS][H]  es energies
constexpr int OFF_WIH0 = OFF_ES   + 2*ROWS*H_*4;          // f32 [G3][2]
constexpr int OFF_BIH0 = OFF_WIH0 + G3*2*4;
constexpr int OFF_BHH0 = OFF_BIH0 + G3*4;
constexpr int OFF_BIH1 = OFF_BHH0 + G3*4;
constexpr int OFF_BHH1 = OFF_BIH1 + G3*4;
constexpr int OFF_FCB  = OFF_BHH1 + G3*4;
constexpr int OFF_VW   = OFF_FCB  + H_*4;                 // f32 [2][H]
constexpr int OFF_XS   = OFF_VW   + 2*H_*4;               // f32 [ROWS][2]
constexpr int OFF_HAH  = OFF_XS   + ROWS*2*4;             // f16 [2][ROWS][H]  hidden mirror (A operand)
constexpr int OFF_HNH  = OFF_HAH  + 2*ROWS*H_*2;          // f16 [2][ROWS][H]  h_new (A operand / buf src)
constexpr int OFF_EHN  = OFF_HNH  + 2*ROWS*H_*2;          // f16 [2][ROWS][H]  current-step eh
constexpr int OFF_HCAT = OFF_EHN  + 2*ROWS*H_*2;          // f16 [2][ROWS][2H] fc input
constexpr int LDS_SZ   = OFF_HCAT + 2*ROWS*H2*2;          // = 169088 bytes
static_assert(LDS_SZ <= 320*1024, "exceeds WGP LDS");
static_assert(OFF_HAH % 16 == 0 && OFF_HNH % 16 == 0 && OFF_EHN % 16 == 0 && OFF_HCAT % 16 == 0, "align");
} // namespace

struct Params {
  const float*    recv;          // [B][T][2]
  const float*    Wih0[2];       // [G3][2]
  const float*    bih0[2], *bhh0[2], *bih1[2], *bhh1[2];   // [G3]
  const _Float16* Whh0[2];       // [G3][H] f16 row-major
  const _Float16* Wih1[2];       // [G3][H]
  const _Float16* Whh1[2];       // [G3][H]
  const _Float16* attn;          // [H][2H]
  const _Float16* fc;            // [H][2H]
  const float*    fc_b;          // [H]
  const float*    v_W;           // [2][H]
  _Float16*       buf[2];        // [NBLK][T][2][ROWS][H] f16
  _Float16*       ehc[2];        // [NBLK][T][2][ROWS][H] f16
};

DEV v8f wmma(v16h a, v16h b, v8f c) {
  return __builtin_amdgcn_wmma_f32_16x16x32_f16(false, a, false, b, (short)0, c, false, false);
}
// A fragment (16x32, 16-bit): lane holds row m=lane&15; q=lane>>4 selects K groups.
// f16 source rows -> two 16-byte contiguous chunks (compiler emits 2x ds_load_b128).
DEV v16h loadA_f16(const _Float16* rp, int k0, int q) {
  v16h a;
  const v8h lo = *(const v8h*)(rp + k0 + q*16);
  const v8h hi = *(const v8h*)(rp + k0 + q*16 + 8);
#pragma unroll
  for (int e = 0; e < 8; ++e) { a[e] = lo[e]; a[e+8] = hi[e]; }
  return a;
}
DEV float sigm(float x) { return 1.f / (1.f + expf(-x)); }

__global__ void cvt_f16_kernel(const float* __restrict__ src, _Float16* __restrict__ dst, int n) {
  int i = blockIdx.x * blockDim.x + threadIdx.x;
  if (i < n) dst[i] = (_Float16)src[i];
}

__global__ __launch_bounds__(THREADS) void gru_attn_kernel(Params P) {
  extern __shared__ __align__(16) char smem[];
  float*    hA    = (float*)(smem + OFF_HA);
  float*    esS   = (float*)(smem + OFF_ES);
  float*    wih0s = (float*)(smem + OFF_WIH0);
  float*    bih0s = (float*)(smem + OFF_BIH0);
  float*    bhh0s = (float*)(smem + OFF_BHH0);
  float*    bih1s = (float*)(smem + OFF_BIH1);
  float*    bhh1s = (float*)(smem + OFF_BHH1);
  float*    fcbs  = (float*)(smem + OFF_FCB);
  float*    vws   = (float*)(smem + OFF_VW);
  float*    xs    = (float*)(smem + OFF_XS);
  _Float16* hAh   = (_Float16*)(smem + OFF_HAH);
  _Float16* hNh   = (_Float16*)(smem + OFF_HNH);
  _Float16* ehN   = (_Float16*)(smem + OFF_EHN);
  _Float16* hcat  = (_Float16*)(smem + OFF_HCAT);

  const int tid = threadIdx.x;
  const int wv  = tid >> 5;
  const int la  = tid & 31;
  const int n   = la & 15;     // WMMA B/C column (and A row)
  const int q   = la >> 4;     // lane half
  const int s   = blockIdx.x >> 4;   // stream
  const int bb  = blockIdx.x & 15;   // batch row-group
  const int r0  = bb * ROWS;

  const _Float16* Whh0 = P.Whh0[s];
  const _Float16* Wih1 = P.Wih1[s];
  const _Float16* Whh1 = P.Whh1[s];
  _Float16* bufS = P.buf[s];
  _Float16* ehcS = P.ehc[s];

  for (int i = tid; i < G3*2; i += THREADS) wih0s[i] = P.Wih0[s][i];
  for (int i = tid; i < G3; i += THREADS) {
    bih0s[i] = P.bih0[s][i]; bhh0s[i] = P.bhh0[s][i];
    bih1s[i] = P.bih1[s][i]; bhh1s[i] = P.bhh1[s][i];
  }
  for (int i = tid; i < H_;   i += THREADS) fcbs[i] = P.fc_b[i];
  for (int i = tid; i < 2*H_; i += THREADS) vws[i]  = P.v_W[i];
  for (int i = tid; i < 2*ROWS*H_; i += THREADS) { hA[i] = 0.f; hAh[i] = (_Float16)0.f; }
  __syncthreads();

  const size_t stepSz  = (size_t)2 * ROWS * H_;       // 8192 elems
  const size_t blkBase = (size_t)bb * T_ * stepSz;

  for (int t = 0; t < T_; ++t) {
    if (tid < ROWS*2)
      xs[tid] = P.recv[(size_t)(r0 + (tid >> 1)) * T_ * 2 + t*2 + (tid & 1)];
    __syncthreads();

    // ---------------- Phase 1: layer-0 GRU (gh = hA0 @ Whh0^T) ----------------
    {
      const int c0 = 16 * wv;
      v8f cr = {}, cz = {}, cn = {};
      const _Float16* rp = hAh + (size_t)(0*ROWS + n) * H_;
      for (int k0 = 0; k0 < H_; k0 += 32) {
        v16h a = loadA_f16(rp, k0, q);
        if (k0 + 32 < H_) __builtin_prefetch(Whh0 + (size_t)(c0 + n)*H_ + k0 + 32, 0, 1);
        v16h br = *(const v16h*)(Whh0 + (size_t)(0*H_ + c0 + n)*H_ + k0 + q*16);
        v16h bz = *(const v16h*)(Whh0 + (size_t)(1*H_ + c0 + n)*H_ + k0 + q*16);
        v16h bn = *(const v16h*)(Whh0 + (size_t)(2*H_ + c0 + n)*H_ + k0 + q*16);
        cr = wmma(a, br, cr); cz = wmma(a, bz, cz); cn = wmma(a, bn, cn);
      }
#pragma unroll
      for (int rr = 0; rr < 8; ++rr) {
        const int m = rr + 8*q, c = c0 + n;
        const float x0 = xs[m*2], x1 = xs[m*2+1];
        const float ir = x0*wih0s[c*2]        + x1*wih0s[c*2+1]        + bih0s[c];
        const float iz = x0*wih0s[(H_+c)*2]   + x1*wih0s[(H_+c)*2+1]   + bih0s[H_+c];
        const float in_= x0*wih0s[(2*H_+c)*2] + x1*wih0s[(2*H_+c)*2+1] + bih0s[2*H_+c];
        const float rg = sigm(ir + cr[rr] + bhh0s[c]);
        const float zg = sigm(iz + cz[rr] + bhh0s[H_+c]);
        const float ng = tanhf(in_ + rg * (cn[rr] + bhh0s[2*H_+c]));
        const float hp = hA[(0*ROWS + m)*H_ + c];
        hNh[(0*ROWS + m)*H_ + c] = (_Float16)((1.f - zg)*ng + zg*hp);
      }
    }
    __syncthreads();

    // ---------------- Phase 2: layer-1 GRU (gi = hN0 @ Wih1^T, gh = hA1 @ Whh1^T) ----
    {
      const int c0 = 16 * wv;
      v8f ir_ = {}, iz_ = {}, in_ = {}, hr_ = {}, hz_ = {}, hn_ = {};
      const _Float16* rp0 = hNh + (size_t)(0*ROWS + n) * H_;
      const _Float16* rp1 = hAh + (size_t)(1*ROWS + n) * H_;
      for (int k0 = 0; k0 < H_; k0 += 32) {
        v16h a0 = loadA_f16(rp0, k0, q);
        v16h a1 = loadA_f16(rp1, k0, q);
        v16h bi0 = *(const v16h*)(Wih1 + (size_t)(0*H_ + c0 + n)*H_ + k0 + q*16);
        v16h bi1 = *(const v16h*)(Wih1 + (size_t)(1*H_ + c0 + n)*H_ + k0 + q*16);
        v16h bi2 = *(const v16h*)(Wih1 + (size_t)(2*H_ + c0 + n)*H_ + k0 + q*16);
        v16h bh0 = *(const v16h*)(Whh1 + (size_t)(0*H_ + c0 + n)*H_ + k0 + q*16);
        v16h bh1 = *(const v16h*)(Whh1 + (size_t)(1*H_ + c0 + n)*H_ + k0 + q*16);
        v16h bh2 = *(const v16h*)(Whh1 + (size_t)(2*H_ + c0 + n)*H_ + k0 + q*16);
        ir_ = wmma(a0, bi0, ir_); iz_ = wmma(a0, bi1, iz_); in_ = wmma(a0, bi2, in_);
        hr_ = wmma(a1, bh0, hr_); hz_ = wmma(a1, bh1, hz_); hn_ = wmma(a1, bh2, hn_);
      }
#pragma unroll
      for (int rr = 0; rr < 8; ++rr) {
        const int m = rr + 8*q, c = c0 + n;
        const float rg = sigm(ir_[rr] + bih1s[c]      + hr_[rr] + bhh1s[c]);
        const float zg = sigm(iz_[rr] + bih1s[H_+c]   + hz_[rr] + bhh1s[H_+c]);
        const float ng = tanhf(in_[rr] + bih1s[2*H_+c] + rg * (hn_[rr] + bhh1s[2*H_+c]));
        const float hp = hA[(1*ROWS + m)*H_ + c];
        hNh[(1*ROWS + m)*H_ + c] = (_Float16)((1.f - zg)*ng + zg*hp);
      }
    }
    __syncthreads();

    // ---------------- Phase 3: buf store + es/eh energy GEMMs ----------------
    {
      _Float16* dst = bufS + blkBase + (size_t)t * stepSz;
      for (int i = tid*8; i < (int)stepSz; i += THREADS*8)
        *(v8h*)(dst + i) = *(const v8h*)(hNh + i);
    }
    {
      const int l = wv >> 3, tb = wv & 7;
      const int g0 = 32*tb, g1 = 32*tb + 16;
      v8f es0 = {}, eh0 = {}, es1 = {}, eh1 = {};
      const _Float16* rp = hNh + (size_t)(l*ROWS + n) * H_;
      for (int k0 = 0; k0 < H_; k0 += 32) {
        v16h a  = loadA_f16(rp, k0, q);
        v16h be0 = *(const v16h*)(P.attn + (size_t)(g0 + n)*H2 +      k0 + q*16);
        v16h bh0 = *(const v16h*)(P.attn + (size_t)(g0 + n)*H2 + H_ + k0 + q*16);
        v16h be1 = *(const v16h*)(P.attn + (size_t)(g1 + n)*H2 +      k0 + q*16);
        v16h bh1 = *(const v16h*)(P.attn + (size_t)(g1 + n)*H2 + H_ + k0 + q*16);
        es0 = wmma(a, be0, es0); eh0 = wmma(a, bh0, eh0);
        es1 = wmma(a, be1, es1); eh1 = wmma(a, bh1, eh1);
      }
#pragma unroll
      for (int rr = 0; rr < 8; ++rr) {
        const int m = rr + 8*q;
        esS[(l*ROWS + m)*H_ + g0 + n] = es0[rr];
        esS[(l*ROWS + m)*H_ + g1 + n] = es1[rr];
        ehN[(l*ROWS + m)*H_ + g0 + n] = (_Float16)eh0[rr];
        ehN[(l*ROWS + m)*H_ + g1 + n] = (_Float16)eh1[rr];
      }
    }
    __syncthreads();
    {
      _Float16* dst = ehcS + blkBase + (size_t)t * stepSz;
      for (int i = tid*8; i < (int)stepSz; i += THREADS*8)
        *(v8h*)(dst + i) = *(const v8h*)(ehN + i);
    }
    __threadfence();
    __syncthreads();

    // ---------------- Phase 4: online-softmax attention (wave = batch row) ----
    {
      const int m = wv;
      float M0 = -1e30f, M1 = -1e30f, S0 = 0.f, S1 = 0.f;
      float acc0[8], acc1[8];
#pragma unroll
      for (int u = 0; u < 8; ++u) { acc0[u] = 0.f; acc1[u] = 0.f; }
      const float* vw0 = vws + 8*la;
      const float* vw1 = vws + H_ + 8*la;
      for (int j = 0; j <= t; ++j) {
#pragma unroll
        for (int l = 0; l < 2; ++l) {
          v8h ehv, bfv;
          if (j == t) {
            ehv = *(const v8h*)(ehN + (size_t)(l*ROWS + m)*H_ + 8*la);
            bfv = *(const v8h*)(hNh + (size_t)(l*ROWS + m)*H_ + 8*la);
          } else {
            const size_t off = blkBase + (size_t)j*stepSz + (size_t)(l*ROWS + m)*H_ + 8*la;
            ehv = *(const v8h*)(ehcS + off);
            bfv = *(const v8h*)(bufS + off);
          }
          const float* esp = esS + (size_t)(l*ROWS + m)*H_ + 8*la;
          float p0 = 0.f, p1 = 0.f;
#pragma unroll
          for (int u = 0; u < 8; ++u) {
            const float th = tanhf(esp[u] + (float)ehv[u]);
            p0 += vw0[u]*th; p1 += vw1[u]*th;
          }
#pragma unroll
          for (int off = 16; off; off >>= 1) {
            p0 += __shfl_xor(p0, off, 32);
            p1 += __shfl_xor(p1, off, 32);
          }
          // online softmax update (o = 0 and o = 1 share the value vector bfv)
          {
            const float Mn = fmaxf(M0, p0);
            const float sc = expf(M0 - Mn), w = expf(p0 - Mn);
            S0 = S0*sc + w; M0 = Mn;
#pragma unroll
            for (int u = 0; u < 8; ++u) acc0[u] = acc0[u]*sc + w*(float)bfv[u];
          }
          {
            const float Mn = fmaxf(M1, p1);
            const float sc = expf(M1 - Mn), w = expf(p1 - Mn);
            S1 = S1*sc + w; M1 = Mn;
#pragma unroll
            for (int u = 0; u < 8; ++u) acc1[u] = acc1[u]*sc + w*(float)bfv[u];
          }
        }
      }
      const float inv0 = 1.f / S0, inv1 = 1.f / S1;
      _Float16* hc0 = hcat + (size_t)(0*ROWS + m)*H2;
      _Float16* hc1 = hcat + (size_t)(1*ROWS + m)*H2;
#pragma unroll
      for (int u = 0; u < 8; ++u) {
        const int h = 8*la + u;
        hc0[h] = (_Float16)(acc0[u] * inv0);
        hc1[h] = (_Float16)(acc1[u] * inv1);
      }
      *(v8h*)(hc0 + H_ + 8*la) = *(const v8h*)(hNh + (size_t)(0*ROWS + m)*H_ + 8*la);
      *(v8h*)(hc1 + H_ + 8*la) = *(const v8h*)(hNh + (size_t)(1*ROWS + m)*H_ + 8*la);
    }
    __syncthreads();

    // ---------------- Phase 5: fc GEMM -> new hidden hA ----------------
    {
      const int l = wv >> 3, tb = wv & 7;
      const int g0 = 32*tb, g1 = 32*tb + 16;
      v8f d0 = {}, d1 = {};
      const _Float16* rp = hcat + (size_t)(l*ROWS + n) * H2;
      for (int k0 = 0; k0 < H2; k0 += 32) {
        v16h a  = loadA_f16(rp, k0, q);
        v16h b0 = *(const v16h*)(P.fc + (size_t)(g0 + n)*H2 + k0 + q*16);
        v16h b1 = *(const v16h*)(P.fc + (size_t)(g1 + n)*H2 + k0 + q*16);
        d0 = wmma(a, b0, d0); d1 = wmma(a, b1, d1);
      }
#pragma unroll
      for (int rr = 0; rr < 8; ++rr) {
        const int m = rr + 8*q;
        const float v0 = d0[rr] + fcbs[g0 + n];
        const float v1 = d1[rr] + fcbs[g1 + n];
        hA [(l*ROWS + m)*H_ + g0 + n] = v0;
        hA [(l*ROWS + m)*H_ + g1 + n] = v1;
        hAh[(l*ROWS + m)*H_ + g0 + n] = (_Float16)v0;
        hAh[(l*ROWS + m)*H_ + g1 + n] = (_Float16)v1;
      }
    }
    __syncthreads();
  }
}

// Final head: out[b,t] = sigmoid(out_W[:H]·buf1[b,t,1,:] + out_W[H:]·buf2[b,idx(t),1,:] + out_b)
__global__ __launch_bounds__(256) void out_head_kernel(const _Float16* __restrict__ buf1,
                                                       const _Float16* __restrict__ buf2,
                                                       const float* __restrict__ outW,
                                                       const float* __restrict__ outB,
                                                       float* __restrict__ out) {
  const int gid = blockIdx.x * 8 + (threadIdx.x >> 5);
  const int la  = threadIdx.x & 31;
  if (gid >= B_ * T_) return;
  const int b = gid >> 6, t = gid & 63;
  const int t2 = (t >= T_ - DLY - 1) ? (T_ - 1) : (t + DLY);
  const int bb = b >> 4, m = b & 15;
  const _Float16* p1 = buf1 + ((((size_t)bb*T_ + t )*2 + 1)*ROWS + m)*H_ + 8*la;
  const _Float16* p2 = buf2 + ((((size_t)bb*T_ + t2)*2 + 1)*ROWS + m)*H_ + 8*la;
  float p = 0.f;
#pragma unroll
  for (int u = 0; u < 8; ++u) {
    const int h = 8*la + u;
    p += outW[h]*(float)p1[u] + outW[H_ + h]*(float)p2[u];
  }
#pragma unroll
  for (int off = 16; off; off >>= 1) p += __shfl_xor(p, off, 32);
  if (la == 0) out[gid] = 1.f / (1.f + expf(-(p + outB[0])));
}

extern "C" void kernel_launch(void* const* d_in, const int* in_sizes, int n_in,
                              void* d_out, int out_size, void* d_ws, size_t ws_size,
                              hipStream_t stream) {
  (void)in_sizes; (void)n_in; (void)out_size; (void)ws_size;

  // ---- carve workspace ----
  char* base = (char*)d_ws;
  size_t off = 0;
  auto alloc = [&](size_t bytes) -> void* {
    void* r = base + off;
    off += (bytes + 255) & ~(size_t)255;
    return r;
  };
  _Float16 *wWhh0[2], *wWih1[2], *wWhh1[2];
  for (int s = 0; s < 2; ++s) {
    wWhh0[s] = (_Float16*)alloc((size_t)G3 * H_ * 2);
    wWih1[s] = (_Float16*)alloc((size_t)G3 * H_ * 2);
    wWhh1[s] = (_Float16*)alloc((size_t)G3 * H_ * 2);
  }
  _Float16* wAttn = (_Float16*)alloc((size_t)H_ * H2 * 2);
  _Float16* wFc   = (_Float16*)alloc((size_t)H_ * H2 * 2);
  _Float16 *buf[2], *ehc[2];
  const size_t bufElems = (size_t)NBLK * T_ * 2 * ROWS * H_;
  for (int s = 0; s < 2; ++s) {
    buf[s] = (_Float16*)alloc(bufElems * 2);
    ehc[s] = (_Float16*)alloc(bufElems * 2);
  }

  // ---- weight conversion f32 -> f16 ----
  auto cvt = [&](const void* src, _Float16* dst, int nel) {
    cvt_f16_kernel<<<(nel + 255) / 256, 256, 0, stream>>>((const float*)src, dst, nel);
  };
  for (int s = 0; s < 2; ++s) {
    const int gb = 1 + 8 * s;   // g1 at 1..8, g2 at 9..16
    cvt(d_in[gb + 1], wWhh0[s], G3 * H_);
    cvt(d_in[gb + 4], wWih1[s], G3 * H_);
    cvt(d_in[gb + 5], wWhh1[s], G3 * H_);
  }
  cvt(d_in[19], wAttn, H_ * H2);   // attn_W
  cvt(d_in[17], wFc,   H_ * H2);   // fc_W

  // ---- main persistent recurrence kernel ----
  Params P;
  P.recv = (const float*)d_in[0];
  for (int s = 0; s < 2; ++s) {
    const int gb = 1 + 8 * s;
    P.Wih0[s] = (const float*)d_in[gb + 0];
    P.bih0[s] = (const float*)d_in[gb + 2];
    P.bhh0[s] = (const float*)d_in[gb + 3];
    P.bih1[s] = (const float*)d_in[gb + 6];
    P.bhh1[s] = (const float*)d_in[gb + 7];
    P.Whh0[s] = wWhh0[s];
    P.Wih1[s] = wWih1[s];
    P.Whh1[s] = wWhh1[s];
    P.buf[s]  = buf[s];
    P.ehc[s]  = ehc[s];
  }
  P.attn = wAttn;
  P.fc   = wFc;
  P.fc_b = (const float*)d_in[18];
  P.v_W  = (const float*)d_in[20];

  gru_attn_kernel<<<dim3(2 * NBLK), dim3(THREADS), LDS_SZ, stream>>>(P);

  // ---- output head ----
  out_head_kernel<<<dim3((B_ * T_) / 8), dim3(256), 0, stream>>>(
      buf[0], buf[1], (const float*)d_in[21], (const float*)d_in[22], (float*)d_out);
}
